// MultiHeadSelfAttention_36507222016352
// MI455X (gfx1250) — matrix operand order repaired and checked by reading, not compiled, not run on hardware
//
#include <hip/hip_runtime.h>
#include <hip/hip_bf16.h>

typedef __bf16 bf16;
typedef bf16  v16bf __attribute__((ext_vector_type(16)));
typedef bf16  v8bf  __attribute__((ext_vector_type(8)));
typedef float v8f   __attribute__((ext_vector_type(8)));
typedef unsigned u32x4 __attribute__((ext_vector_type(4)));
typedef int      i32x8 __attribute__((ext_vector_type(8)));
typedef int      i32x4 __attribute__((ext_vector_type(4)));

__device__ __forceinline__ bf16 f2bf(float f) {
  union { float f; unsigned u; } v; v.f = f;
  unsigned r = v.u + 0x7FFFu + ((v.u >> 16) & 1u);   // round-to-nearest-even
  unsigned short h = (unsigned short)(r >> 16);
  union { unsigned short s; bf16 b; } o; o.s = h;
  return o.b;
}

// Flat pointer to a __shared__ object: low 32 bits are the LDS byte offset.
__device__ __forceinline__ unsigned ldsoff(const void* p) {
  return (unsigned)(uintptr_t)p;
}

// Async global->LDS 16B copy (per-lane), tracked with ASYNCcnt.
__device__ __forceinline__ void async_cp16(unsigned lds_byte, const void* gptr) {
  asm volatile("global_load_async_to_lds_b128 %0, %1, off"
               :: "v"(lds_byte), "v"(gptr) : "memory");
}
__device__ __forceinline__ void wait_async0() {
  asm volatile("s_wait_asynccnt 0" ::: "memory");
}

#if __has_builtin(__builtin_amdgcn_tensor_load_to_lds)
#define HAVE_TDM 1
// TDM: load a 64x64 bf16 tile (row stride 1024 elems) into LDS with rows padded
// to 72 elems (pad 4 DWORDs after every 32 DWORDs).
__device__ __forceinline__ void tdm_load_tile64(unsigned lds_byte, const bf16* gtile) {
  unsigned long long ga = (unsigned long long)(uintptr_t)gtile;
  u32x4 g0;
  g0.x = 1u;                                             // count=1, user mode
  g0.y = lds_byte;                                       // lds_addr
  g0.z = (unsigned)(ga & 0xFFFFFFFFu);                   // global_addr[31:0]
  g0.w = (unsigned)((ga >> 32) & 0x01FFFFFFu) | 0x80000000u; // addr[56:32], type=2
  i32x8 g1;
  g1[0] = (1 << 16)       // data_size = 2 bytes
        | (1 << 20)       // pad_enable
        | (4 << 22)       // pad_interval: 2^(4+1) = 32 DWORDs (one 128B row)
        | (3 << 25);      // pad_amount: 4 DWORDs (16B) -> 144B LDS rows
  g1[1] = (int)(1024u << 16);   // tensor_dim0 = 1024 (low 16)
  g1[2] = (int)(4096u << 16);   // tensor_dim0 hi=0 | tensor_dim1 = 4096 (low 16)
  g1[3] = (int)(64u << 16);     // tensor_dim1 hi=0 | tile_dim0 = 64
  g1[4] = 64;                   // tile_dim1 = 64, tile_dim2 = 0
  g1[5] = 1024;                 // tensor_dim0_stride = 1024 elems
  g1[6] = 0;
  g1[7] = 0;
  i32x4 z4 = {0, 0, 0, 0};
  i32x8 z8 = {0, 0, 0, 0, 0, 0, 0, 0};
  __builtin_amdgcn_tensor_load_to_lds(g0, g1, z4, z4, z8, 0);  // 6-arg toolchain
}
#endif

// A-fragment (16x32 bf16, MxK): lane m = lane&15, half hb = lane>>4.
__device__ __forceinline__ v16bf load_frag_a(const bf16* row, int kbase) {
  v8bf lo = *(const v8bf*)(row + kbase);
  v8bf hi = *(const v8bf*)(row + kbase + 8);
  return __builtin_shufflevector(lo, hi, 0,1,2,3,4,5,6,7,8,9,10,11,12,13,14,15);
}
// B-fragment (32x16 bf16, KxN): lane n = lane&15, 16 contiguous K at (lane>>4)*16
__device__ __forceinline__ v16bf load_frag_b(const bf16* p) {
  v8bf lo = *(const v8bf*)(p);
  v8bf hi = *(const v8bf*)(p + 8);
  return __builtin_shufflevector(lo, hi, 0,1,2,3,4,5,6,7,8,9,10,11,12,13,14,15);
}
__device__ __forceinline__ v8f wmma_bf(v16bf a, v16bf b, v8f c) {
  return __builtin_amdgcn_wmma_f32_16x16x32_bf16(false, a, false, b, (short)0, c, false, false);
}

// ---------------------------------------------------------------- convert
__global__ void cvt_f32_bf16(const float* __restrict__ in, bf16* __restrict__ out, int n) {
  int i = blockIdx.x * blockDim.x + threadIdx.x;
  if (i < n) out[i] = f2bf(in[i]);
}

// ---------------------------------------------------------------- GEMM  C = A * W^T
// A: [M,1024] bf16, W: [N,1024] bf16, C: [M,N] fp32.
// 256 thr (8 waves, 2x4). Block tile 64(M) x 256(N); wave tile 32x64.
// Double-buffered LDS fed by async global->LDS copies.
#define GEMM_K 1024
__global__ __launch_bounds__(256)
void gemm_xwT(const bf16* __restrict__ A, const bf16* __restrict__ W,
              float* __restrict__ C, int M, int N) {
  const int LDA = 40, LDB = 40;                 // 32 + 8 bf16 pad
  __shared__ bf16 As[2][64 * 40];
  __shared__ bf16 Bs[2][256 * 40];
  int t = threadIdx.x;
  int wave = t >> 5, lane = t & 31;
  int wm = wave & 1, wn = wave >> 1;            // 2 x 4 wave grid
  int m0 = blockIdx.x * 64, n0 = blockIdx.y * 256;
  int hb = lane >> 4, lm = lane & 15;
  v8f c[2][4] = {};

  int arow = t >> 2, acol = (t & 3) * 8;        // A stage: 64 rows x 4 groups

  // stage tile 0 into buffer 0 (async)
  {
    async_cp16(ldsoff(&As[0][arow * LDA + acol]),
               &A[(size_t)(m0 + arow) * GEMM_K + acol]);
#pragma unroll
    for (int rep = 0; rep < 4; ++rep) {
      int idx = t + rep * 256;
      int brow = idx >> 2, bcol = (idx & 3) * 8;
      async_cp16(ldsoff(&Bs[0][brow * LDB + bcol]),
                 &W[(size_t)(n0 + brow) * GEMM_K + bcol]);
    }
  }

  int cur = 0;
  for (int k0 = 0; k0 < GEMM_K; k0 += 32) {
    wait_async0();
    __syncthreads();            // tile k visible; everyone done with buffer cur^1
    if (k0 + 32 < GEMM_K) {     // async-prefetch tile k+1 into the other buffer
      int nb = cur ^ 1, kn = k0 + 32;
      async_cp16(ldsoff(&As[nb][arow * LDA + acol]),
                 &A[(size_t)(m0 + arow) * GEMM_K + kn + acol]);
#pragma unroll
      for (int rep = 0; rep < 4; ++rep) {
        int idx = t + rep * 256;
        int brow = idx >> 2, bcol = (idx & 3) * 8;
        async_cp16(ldsoff(&Bs[nb][brow * LDB + bcol]),
                   &W[(size_t)(n0 + brow) * GEMM_K + kn + bcol]);
      }
    }
#pragma unroll
    for (int i = 0; i < 2; ++i) {
      v16bf a = load_frag_a(&As[cur][(wm * 32 + i * 16 + lm) * LDA], hb * 16);
#pragma unroll
      for (int j = 0; j < 4; ++j) {
        v16bf b = load_frag_b(&Bs[cur][(wn * 64 + j * 16 + lm) * LDB + hb * 16]);
        c[i][j] = wmma_bf(a, b, c[i][j]);
      }
    }
    cur ^= 1;
  }
  // C layout: VGPR r -> row (r + 8*hb), col = lm
#pragma unroll
  for (int i = 0; i < 2; ++i)
#pragma unroll
    for (int j = 0; j < 4; ++j)
#pragma unroll
      for (int r = 0; r < 8; ++r) {
        int row = m0 + wm * 32 + i * 16 + r + 8 * hb;
        int col = n0 + wn * 64 + j * 16 + lm;
        C[(size_t)row * N + col] = c[i][j][r];
      }
}

// ---------------------------------------------------------------- RoPE + bf16
__global__ void rope_cvt(const float* __restrict__ q32, const float* __restrict__ k32,
                         const float* __restrict__ v32, const int* __restrict__ pos,
                         bf16* __restrict__ qbf, bf16* __restrict__ kbf,
                         bf16* __restrict__ vbf) {
  int idx = blockIdx.x * blockDim.x + threadIdx.x;  // B*S*512 threads exactly
  int row = idx >> 9;
  int pr  = idx & 511;
  int h = pr >> 5, i = pr & 31;
  int s = row & 2047;
  float p = (float)pos[s];
  float inv = __expf((float)i * (-2.0f / 64.0f) * 9.210340371976184f); // 10000^(-2i/64)
  float ang = p * inv;
  float cs = __cosf(ang), sn = __sinf(ang);
  size_t base = (size_t)row * 1024 + h * 64 + 2 * i;
  float q1 = q32[base], q2 = q32[base + 1];
  float k1 = k32[base], k2 = k32[base + 1];
  const float qs = 0.125f;                           // 1/sqrt(Dh)
  qbf[base]     = f2bf((q1 * cs - q2 * sn) * qs);
  qbf[base + 1] = f2bf((q1 * sn + q2 * cs) * qs);
  kbf[base]     = f2bf(k1 * cs - k2 * sn);
  kbf[base + 1] = f2bf(k1 * sn + k2 * cs);
  vbf[base]     = f2bf(v32[base]);
  vbf[base + 1] = f2bf(v32[base + 1]);
}

// ---------------------------------------------------------------- flash attention
// grid (S/64, H, B), 128 threads (4 waves). Wave w owns query rows [qb*64+w*16, +16).
__global__ __launch_bounds__(128)
void flash_attn(const bf16* __restrict__ Q, const bf16* __restrict__ K,
                const bf16* __restrict__ V, bf16* __restrict__ O) {
  const int LD = 72;                       // 64 + 8 pad (matches TDM pad descriptor)
  __shared__ bf16 Qt[64 * 72];
  __shared__ bf16 Kt[64 * 72];             // [key][dh]
  __shared__ bf16 Vt[64 * 72];             // [dh][key] (transposed)
  __shared__ bf16 Pt[4 * 16 * 72];         // per-wave P staging (C->A relayout)
  int qb = blockIdx.x, h = blockIdx.y, b = blockIdx.z;
  int t = threadIdx.x, wave = t >> 5, lane = t & 31;
  int hb = lane >> 4, lm = lane & 15;
  size_t rowbase = (size_t)b * 2048;
  int col0 = h * 64;

  // ---- stage Q tile (64 x 64)
#ifdef HAVE_TDM
  if (wave == 0) {
    tdm_load_tile64(ldsoff(Qt), &Q[(rowbase + qb * 64) * 1024 + col0]);
    __builtin_amdgcn_s_wait_tensorcnt(0);
  }
#else
  {
    int r = t >> 1, cb = (t & 1) * 32;
    const bf16* src = &Q[(rowbase + qb * 64 + r) * 1024 + col0 + cb];
#pragma unroll
    for (int u = 0; u < 4; ++u)
      async_cp16(ldsoff(&Qt[r * LD + cb + u * 8]), src + u * 8);
    wait_async0();
  }
#endif

  v8f o[4] = {};
  float mrun[8], lrun[8];
#pragma unroll
  for (int r = 0; r < 8; ++r) { mrun[r] = -1e30f; lrun[r] = 0.f; }

  __syncthreads();

  for (int kb = 0; kb <= qb; ++kb) {
    // ---- stage K tile (TDM DMA) overlapped with manual V transpose
#ifdef HAVE_TDM
    if (wave == 0)
      tdm_load_tile64(ldsoff(Kt), &K[(rowbase + kb * 64) * 1024 + col0]);
#else
    {
      int r = t >> 1, cb = (t & 1) * 32;
      const bf16* ks = &K[(rowbase + kb * 64 + r) * 1024 + col0 + cb];
#pragma unroll
      for (int u = 0; u < 4; ++u)
        async_cp16(ldsoff(&Kt[r * LD + cb + u * 8]), ks + u * 8);
    }
#endif
    {
      int r = t >> 1, cb = (t & 1) * 32;
      const bf16* vs = &V[(rowbase + kb * 64 + r) * 1024 + col0 + cb];
      __builtin_prefetch(vs + 64 * 1024, 0, 1);
#pragma unroll
      for (int u = 0; u < 4; ++u) {
        v8bf vv = *(const v8bf*)(vs + u * 8);
#pragma unroll
        for (int e = 0; e < 8; ++e)
          Vt[(cb + u * 8 + e) * LD + r] = vv[e];
      }
    }
#ifdef HAVE_TDM
    if (wave == 0) __builtin_amdgcn_s_wait_tensorcnt(0);
#else
    wait_async0();
#endif
    __syncthreads();

    // ---- S = Q * K^T  (16x64 per wave; K-dim = Dh = 2 chunks of 32)
    v8f s[4] = {};
#pragma unroll
    for (int kc = 0; kc < 2; ++kc) {
      v16bf aq = load_frag_a(&Qt[(wave * 16 + lm) * LD], kc * 32 + hb * 16);
#pragma unroll
      for (int g = 0; g < 4; ++g) {
        v16bf bk = load_frag_b(&Kt[(g * 16 + lm) * LD + kc * 32 + hb * 16]);
        s[g] = wmma_bf(aq, bk, s[g]);
      }
    }

    // ---- causal mask on diagonal block
    if (kb == qb) {
#pragma unroll
      for (int g = 0; g < 4; ++g)
#pragma unroll
        for (int r = 0; r < 8; ++r) {
          int key = kb * 64 + g * 16 + lm;
          int qr  = qb * 64 + wave * 16 + r + 8 * hb;
          if (key > qr) s[g][r] = -1e30f;
        }
    }

    // ---- online softmax (rows span 16-lane halves; xor masks 1/2/4/8)
#pragma unroll
    for (int r = 0; r < 8; ++r) {
      float mx = s[0][r];
#pragma unroll
      for (int g = 1; g < 4; ++g) mx = fmaxf(mx, s[g][r]);
#pragma unroll
      for (int off = 1; off < 16; off <<= 1)
        mx = fmaxf(mx, __shfl_xor(mx, off, 32));
      float mnew = fmaxf(mrun[r], mx);
      float corr = __expf(mrun[r] - mnew);
      float rs = 0.f;
#pragma unroll
      for (int g = 0; g < 4; ++g) {
        float p = __expf(s[g][r] - mnew);
        s[g][r] = p;
        rs += p;
      }
#pragma unroll
      for (int off = 1; off < 16; off <<= 1)
        rs += __shfl_xor(rs, off, 32);
      lrun[r] = lrun[r] * corr + rs;
      mrun[r] = mnew;
#pragma unroll
      for (int d = 0; d < 4; ++d) o[d][r] *= corr;
    }

    // ---- stage P (C-layout -> LDS -> A-layout), per-wave region
#pragma unroll
    for (int g = 0; g < 4; ++g)
#pragma unroll
      for (int r = 0; r < 8; ++r)
        Pt[(wave * 16 + r + 8 * hb) * LD + g * 16 + lm] = f2bf(s[g][r]);

    // ---- O += P * V   (K-dim = 64 keys = 2 chunks of 32)
#pragma unroll
    for (int kc = 0; kc < 2; ++kc) {
      v16bf ap = load_frag_a(&Pt[(wave * 16 + lm) * LD], kc * 32 + hb * 16);
#pragma unroll
      for (int d = 0; d < 4; ++d) {
        v16bf bv = load_frag_b(&Vt[(d * 16 + lm) * LD + kc * 32 + hb * 16]);
        o[d] = wmma_bf(ap, bv, o[d]);
      }
    }
    __syncthreads();
  }

  // ---- normalize and store bf16
#pragma unroll
  for (int r = 0; r < 8; ++r) {
    float inv = 1.0f / lrun[r];
    int row = qb * 64 + wave * 16 + r + 8 * hb;
    size_t addr = (rowbase + row) * 1024 + col0;
#pragma unroll
    for (int d = 0; d < 4; ++d)
      O[addr + d * 16 + lm] = f2bf(o[d][r] * inv);
  }
}

// ---------------------------------------------------------------- launch
extern "C" void kernel_launch(void* const* d_in, const int* in_sizes, int n_in,
                              void* d_out, int out_size, void* d_ws, size_t ws_size,
                              hipStream_t stream) {
  const float* x  = (const float*)d_in[0];
  const int*  pos = (const int*)d_in[1];
  const float* Wq = (const float*)d_in[2];
  const float* Wk = (const float*)d_in[3];
  const float* Wv = (const float*)d_in[4];
  const float* Wo = (const float*)d_in[5];
  float* out = (float*)d_out;

  const size_t NT = 4096;  // B*S tokens
  const size_t D  = 1024;

  char* ws = (char*)d_ws;
  bf16*  xbf  = (bf16*)ws;  ws += NT * D * 2;
  bf16*  wqbf = (bf16*)ws;  ws += D * D * 2;
  bf16*  wkbf = (bf16*)ws;  ws += D * D * 2;
  bf16*  wvbf = (bf16*)ws;  ws += D * D * 2;
  bf16*  wobf = (bf16*)ws;  ws += D * D * 2;
  float* q32  = (float*)ws; ws += NT * D * 4;
  float* k32  = (float*)ws; ws += NT * D * 4;
  float* v32  = (float*)ws; ws += NT * D * 4;
  bf16*  qbf  = (bf16*)ws;  ws += NT * D * 2;
  bf16*  kbf  = (bf16*)ws;  ws += NT * D * 2;
  bf16*  vbf  = (bf16*)ws;  ws += NT * D * 2;
  bf16*  abf  = (bf16*)ws;  ws += NT * D * 2;

  int nx = (int)(NT * D), nw = (int)(D * D);
  cvt_f32_bf16<<<(nx + 255) / 256, 256, 0, stream>>>(x,  xbf,  nx);
  cvt_f32_bf16<<<(nw + 255) / 256, 256, 0, stream>>>(Wq, wqbf, nw);
  cvt_f32_bf16<<<(nw + 255) / 256, 256, 0, stream>>>(Wk, wkbf, nw);
  cvt_f32_bf16<<<(nw + 255) / 256, 256, 0, stream>>>(Wv, wvbf, nw);
  cvt_f32_bf16<<<(nw + 255) / 256, 256, 0, stream>>>(Wo, wobf, nw);

  dim3 gg(64, 4);  // 4096/64 x 1024/256
  gemm_xwT<<<gg, 256, 0, stream>>>(xbf, wqbf, q32, 4096, 1024);
  gemm_xwT<<<gg, 256, 0, stream>>>(xbf, wkbf, k32, 4096, 1024);
  gemm_xwT<<<gg, 256, 0, stream>>>(xbf, wvbf, v32, 4096, 1024);

  rope_cvt<<<8192, 256, 0, stream>>>(q32, k32, v32, pos, qbf, kbf, vbf);

  flash_attn<<<dim3(32, 16, 2), 128, 0, stream>>>(qbf, kbf, vbf, abf);

  gemm_xwT<<<gg, 256, 0, stream>>>(abf, wobf, out, 4096, 1024);
}
